// GraphLearner_54949811585484
// MI455X (gfx1250) — matrix operand order repaired and checked by reading, not compiled, not run on hardware
//
#include <hip/hip_runtime.h>
#include <hip/hip_bf16.h>
#include <stdint.h>

// Problem constants (from reference)
#define B_    16
#define CLS_  2048
#define D_    256
#define N_    (CLS_ + 1)     // 2049
#define NCT_  129            // column tiles swept (129*16 = 2064 >= N)
#define NRBLK_ 33            // row blocks (each covers 4 row tiles = 64 rows)
#define NPAD_ (NRBLK_ * 64)  // 2112 padded rows
#define ALPHA_IT 0.7f
#define BETA_IT  0.5f

typedef __attribute__((ext_vector_type(16))) __bf16 v16bf;
typedef __attribute__((ext_vector_type(8)))  float  v8f;

union Frag {
    uint4 u[2];   // 2 x 16B = 16 bf16
    v16bf v;
};

__device__ __forceinline__ uint16_t f2bf(float f) {
    uint32_t x = __float_as_uint(f);
    uint32_t r = x + 0x7FFFu + ((x >> 16) & 1u);   // RNE
    return (uint16_t)(r >> 16);
}
__device__ __forceinline__ float bf2f(uint16_t h) {
    return __uint_as_float(((uint32_t)h) << 16);
}

// ---------------------------------------------------------------------------
// 1) Build concat(input, base) row n, L2-normalize, apply row mask, store bf16
//    into padded (B, NPAD, D) buffer. Pad rows (n >= N) are zero.
// ---------------------------------------------------------------------------
__global__ void prep_xn(const float* __restrict__ inp,    // (B,1,D)
                        const float* __restrict__ base,   // (B,CLS,D)
                        const float* __restrict__ mask,   // (B,N)
                        uint16_t*    __restrict__ xn)     // (B,NPAD,D) bf16
{
    int n = blockIdx.x;        // 0..NPAD-1
    int b = blockIdx.y;
    int d = threadIdx.x;       // 0..255
    __shared__ float red[D_];

    float x = 0.0f;
    if (n < N_) {
        x = (n == 0) ? inp[b * D_ + d]
                     : base[((long)b * CLS_ + (n - 1)) * D_ + d];
    }
    red[d] = x * x;
    __syncthreads();
    for (int s = D_ / 2; s > 0; s >>= 1) {
        if (d < s) red[d] += red[d + s];
        __syncthreads();
    }
    float nrm = sqrtf(red[0]);
    float scl = (n < N_) ? mask[b * N_ + n] / fmaxf(nrm, 1e-12f) : 0.0f;
    xn[((long)b * NPAD_ + n) * D_ + d] = f2bf(x * scl);
}

// ---------------------------------------------------------------------------
// 2) Gram relu row-sums via bf16 WMMA:  rowsum[b,n] = sum_k relu(Xn·Xnᵀ)[n,k]
//    Block = 4 waves, each wave owns one 16-row tile (64 rows per block).
//    All waves sweep the same column tiles; the 16x256 B panel is staged in
//    LDS (double-buffered, bank-padded rows). Panels are fetched with the
//    CDNA5 async DMA path (global_load_async_to_lds_b128, ASYNCcnt) issued
//    one tile ahead so the transfer overlaps the 8 chained WMMAs, with no
//    VGPR staging at all.
// ---------------------------------------------------------------------------
#define LROW 264   // bf16 elements per LDS row incl. 16B pad (528B -> banks +4/row)

// Issue this thread's share (4 x b128) of one 16x256 bf16 panel:
//   global tile t (rows t*16 .. t*16+15 of Xb) -> LDS at lbase.
__device__ __forceinline__ void issue_tile_async(const uint16_t* gbase,
                                                 uint32_t lbase, int t)
{
#pragma unroll
    for (int i = 0; i < 4; ++i) {
        uint32_t loff = lbase + (uint32_t)(i * 128);          // bytes
        const uint16_t* g = gbase + (size_t)t * (16 * D_) + i * 64;
        asm volatile("global_load_async_to_lds_b128 %0, %1, off"
                     :: "v"(loff), "v"(g)
                     : "memory");
    }
}
__device__ __forceinline__ void wait_async0() {
    asm volatile("s_wait_asynccnt 0x0" ::: "memory");
}

__global__ void __launch_bounds__(128)
gram_rowsum(const uint16_t* __restrict__ xn,     // (B,NPAD,D) bf16
            float*          __restrict__ rowsum) // (B,NPAD) pre-zeroed
{
    const int b    = blockIdx.y;
    const int wave = threadIdx.x >> 5;
    const int lane = threadIdx.x & 31;
    const int l15  = lane & 15;
    const int hi   = (lane >> 4) & 1;
    const int rt   = blockIdx.x * 4 + wave;      // this wave's row tile (< 132)

    const uint16_t* Xb = xn + (long)b * NPAD_ * D_;

    __shared__ __align__(16) uint16_t lbuf[2][16 * LROW];  // 2 x 8448 B

    // Cooperative panel mapping: 128 threads, 16 rows x 32 uint4 (4 per thread).
    const int sr = threadIdx.x >> 3;   // row 0..15
    const int sc = threadIdx.x & 7;    // 8 threads per row

    // Per-thread DMA base addresses (LDS byte address = flat addr low 32 bits).
    const uint16_t* gbase = Xb + (size_t)sr * D_ + sc * 8;
    const uint32_t  lb0 = (uint32_t)(uintptr_t)&lbuf[0][sr * LROW + sc * 8];
    const uint32_t  lb1 = (uint32_t)(uintptr_t)&lbuf[1][sr * LROW + sc * 8];

    // --- Hoist the 8 A-fragments for this wave's 16-row strip (K = 0..255).
    // 16-bit A layout: lane -> M = lane&15; K chunks split by lane half.
    const uint16_t* arow = Xb + (rt * 16 + l15) * D_;
    Frag a[8];
#pragma unroll
    for (int kk = 0; kk < 8; ++kk) {
        const uint16_t* p = arow + kk * 32 + hi * 8;
        a[kk].u[0] = *(const uint4*)(p);        // K = kk*32 + hi*8 + [0..7]
        a[kk].u[1] = *(const uint4*)(p + 16);   // K = kk*32 + 16 + hi*8 + [0..7]
    }

    float part[8] = {0.f, 0.f, 0.f, 0.f, 0.f, 0.f, 0.f, 0.f};

    // Prologue: DMA tile 0 -> buf0, make visible, then start tile 1 -> buf1.
    issue_tile_async(gbase, lb0, 0);
    wait_async0();
    __syncthreads();
    issue_tile_async(gbase, lb1, 1);

    for (int ct = 0; ct < NCT_; ++ct) {
        const int cur = ct & 1;

        // 8 chained WMMAs on the LDS-resident B panel.
        v8f acc = {0.f, 0.f, 0.f, 0.f, 0.f, 0.f, 0.f, 0.f};
#pragma unroll
        for (int kk = 0; kk < 8; ++kk) {
            Frag bf;
            const uint16_t* bp = &lbuf[cur][l15 * LROW + kk * 32 + hi * 8];
            bf.u[0] = *(const uint4*)(bp);
            bf.u[1] = *(const uint4*)(bp + 16);
            acc = __builtin_amdgcn_wmma_f32_16x16x32_bf16(
                      false, a[kk].v, false, bf.v, (short)0, acc, false, false);
        }
#pragma unroll
        for (int r = 0; r < 8; ++r) part[r] += fmaxf(acc[r], 0.0f);

        // Tile ct+1 (in flight -> lbuf[cur^1]): wait my wave's share, then
        // barrier so (a) every wave's share is visible and (b) everyone is
        // done reading lbuf[cur] before we DMA tile ct+2 into it.
        wait_async0();
        __syncthreads();
        if (ct + 2 < NCT_)
            issue_tile_async(gbase, cur ? lb1 : lb0, ct + 2);
    }

    // Reduce across the 16 lanes of each half (sum over columns of the tile).
#pragma unroll
    for (int m = 1; m <= 8; m <<= 1) {
#pragma unroll
        for (int r = 0; r < 8; ++r) part[r] += __shfl_xor(part[r], m, 32);
    }

    if (l15 == 0) {
        int rowbase = rt * 16 + hi * 8;   // VGPR r of half hi = row rowbase + r
#pragma unroll
        for (int r = 0; r < 8; ++r)
            atomicAdd(&rowsum[(long)b * NPAD_ + rowbase + r], part[r]);
    }
}

// ---------------------------------------------------------------------------
// 3) Normalized adjacency row 0:
//    w[b,m] = d0 * (relu(G[0,m]) + (m==0)) * mask[b,m] * d_m
//    with d_n = rsqrt(rowsum[b,n] + mask[b,n] + 1e-8)   (+mask = identity term)
// ---------------------------------------------------------------------------
__global__ void weights_k(const uint16_t* __restrict__ xn,
                          const float*    __restrict__ mask,
                          const float*    __restrict__ rowsum,
                          float*          __restrict__ w)  // (B,NPAD)
{
    int b = blockIdx.y;
    int m = blockIdx.x * 256 + threadIdx.x;
    __shared__ float x0[D_];
    const uint16_t* X0 = xn + (long)b * NPAD_ * D_;
    x0[threadIdx.x] = bf2f(X0[threadIdx.x]);   // node-0 row
    __syncthreads();
    if (m >= NPAD_) return;

    float g = 0.0f;
    const uint16_t* xm = X0 + (long)m * D_;
    for (int d = 0; d < D_; ++d) g += x0[d] * bf2f(xm[d]);

    float val = 0.0f;
    if (m < N_) {
        float mk = mask[(long)b * N_ + m];
        float d0 = rsqrtf(rowsum[(long)b * NPAD_] + mask[(long)b * N_] + 1e-8f);
        float dm = rsqrtf(rowsum[(long)b * NPAD_ + m] + mk + 1e-8f);
        float A0m = (fmaxf(g, 0.0f) + (m == 0 ? 1.0f : 0.0f)) * mk;
        val = d0 * A0m * dm;
    }
    w[(long)b * NPAD_ + m] = val;
}

// ---------------------------------------------------------------------------
// 4) v[b,:] = sum_m w[b,m] * feat[b,m,:]   (feat = concat(input, base), fp32)
// ---------------------------------------------------------------------------
#define VPARTS 16
#define VCHUNK 132   // VPARTS*VCHUNK = 2112 >= N
__global__ void vmix_k(const float* __restrict__ inp,
                       const float* __restrict__ base,
                       const float* __restrict__ w,
                       float*       __restrict__ v)     // (B,D) pre-zeroed
{
    int b = blockIdx.y;
    int part = blockIdx.x;
    int d = threadIdx.x;
    float acc = 0.0f;
    for (int i = 0; i < VCHUNK; ++i) {
        int m = part * VCHUNK + i;
        if (m >= N_) break;
        float wm = w[(long)b * NPAD_ + m];
        float x = (m == 0) ? inp[b * D_ + d]
                           : base[((long)b * CLS_ + (m - 1)) * D_ + d];
        acc += wm * x;
    }
    atomicAdd(&v[b * D_ + d], acc);
}

// ---------------------------------------------------------------------------
// 5) graph0[b,e] = a*tanh(v_tt@W_tt + b_tt[0])*tm0 + (1-a)*tanh(v_it@W_it + b_it[0])*im0
// ---------------------------------------------------------------------------
__global__ void g0_k(const float* __restrict__ vtt, const float* __restrict__ vit,
                     const float* __restrict__ Wtt, const float* __restrict__ btt,
                     const float* __restrict__ Wit, const float* __restrict__ bit,
                     const float* __restrict__ tmask, const float* __restrict__ imask,
                     float* __restrict__ graph0)
{
    int b = blockIdx.x, e = threadIdx.x;
    float stt = 0.0f, sit = 0.0f;
    for (int d = 0; d < D_; ++d) {
        stt += vtt[b * D_ + d] * Wtt[d * D_ + e];
        sit += vit[b * D_ + d] * Wit[d * D_ + e];
    }
    float gtt = tanhf(stt + btt[e]) * tmask[(long)b * N_];
    float git = tanhf(sit + bit[e]) * imask[(long)b * N_];
    graph0[b * D_ + e] = ALPHA_IT * gtt + (1.0f - ALPHA_IT) * git;
}

// ---------------------------------------------------------------------------
// 6) Output blend (masks indexed [:, :CLS] per reference)
// ---------------------------------------------------------------------------
__global__ void fuse_k(const float* __restrict__ bt, const float* __restrict__ bi,
                       const float* __restrict__ tmask, const float* __restrict__ imask,
                       const float* __restrict__ graph0, float* __restrict__ out)
{
    long idx = (long)blockIdx.x * blockDim.x + threadIdx.x;
    const long tot = (long)B_ * CLS_ * D_;
    if (idx >= tot) return;
    int e = (int)(idx % D_);
    long bc = idx / D_;
    int c = (int)(bc % CLS_);
    int b = (int)(bc / CLS_);
    float g = graph0[b * D_ + e];
    out[idx]       = (BETA_IT * bt[idx] + (1.0f - BETA_IT) * g) * tmask[(long)b * N_ + c];
    out[tot + idx] = (BETA_IT * bi[idx] + (1.0f - BETA_IT) * g) * imask[(long)b * N_ + c];
}

// ---------------------------------------------------------------------------
extern "C" void kernel_launch(void* const* d_in, const int* in_sizes, int n_in,
                              void* d_out, int out_size, void* d_ws, size_t ws_size,
                              hipStream_t stream) {
    const float* input_text = (const float*)d_in[0];
    const float* input_img  = (const float*)d_in[1];
    const float* base_text  = (const float*)d_in[2];
    const float* base_img   = (const float*)d_in[3];
    const float* text_mask  = (const float*)d_in[4];
    const float* img_mask   = (const float*)d_in[5];
    const float* W_tt       = (const float*)d_in[6];
    const float* b_tt       = (const float*)d_in[7];
    const float* W_it       = (const float*)d_in[8];
    const float* b_it       = (const float*)d_in[9];
    float* out = (float*)d_out;

    // Workspace layout
    const long xn_elems = (long)B_ * NPAD_ * D_;
    const long rs_elems = (long)B_ * NPAD_;
    const long v_elems  = (long)B_ * D_;
    uint16_t* xn_tt = (uint16_t*)d_ws;
    uint16_t* xn_it = xn_tt + xn_elems;
    float* rs_tt  = (float*)(xn_it + xn_elems);
    float* rs_it  = rs_tt + rs_elems;
    float* w_tt   = rs_it + rs_elems;
    float* w_it   = w_tt + rs_elems;
    float* v_tt   = w_it + rs_elems;
    float* v_it   = v_tt + v_elems;
    float* graph0 = v_it + v_elems;

    // Zero accumulators (rowsums + v); w/graph0 are fully overwritten.
    hipMemsetAsync(rs_tt, 0, (size_t)(2 * rs_elems) * sizeof(float), stream);
    hipMemsetAsync(v_tt,  0, (size_t)(2 * v_elems) * sizeof(float), stream);

    prep_xn<<<dim3(NPAD_, B_), 256, 0, stream>>>(input_text, base_text, text_mask, xn_tt);
    prep_xn<<<dim3(NPAD_, B_), 256, 0, stream>>>(input_img,  base_img,  img_mask,  xn_it);

    gram_rowsum<<<dim3(NRBLK_, B_), 128, 0, stream>>>(xn_tt, rs_tt);
    gram_rowsum<<<dim3(NRBLK_, B_), 128, 0, stream>>>(xn_it, rs_it);

    weights_k<<<dim3((NPAD_ + 255) / 256, B_), 256, 0, stream>>>(xn_tt, text_mask, rs_tt, w_tt);
    weights_k<<<dim3((NPAD_ + 255) / 256, B_), 256, 0, stream>>>(xn_it, img_mask,  rs_it, w_it);

    vmix_k<<<dim3(VPARTS, B_), 256, 0, stream>>>(input_text, base_text, w_tt, v_tt);
    vmix_k<<<dim3(VPARTS, B_), 256, 0, stream>>>(input_img,  base_img,  w_it, v_it);

    g0_k<<<B_, 256, 0, stream>>>(v_tt, v_it, W_tt, b_tt, W_it, b_it,
                                 text_mask, img_mask, graph0);

    const long tot = (long)B_ * CLS_ * D_;
    fuse_k<<<(int)((tot + 255) / 256), 256, 0, stream>>>(base_text, base_img,
                                                         text_mask, img_mask,
                                                         graph0, out);
}
